// RNNWrapper_89489938579723
// MI455X (gfx1250) — matrix operand order repaired and checked by reading, not compiled, not run on hardware
//
#include <hip/hip_runtime.h>
#include <stdint.h>

// ---- CDNA5 WMMA types -------------------------------------------------------
typedef __bf16 bf16_t;
typedef bf16_t v16bf  __attribute__((ext_vector_type(16)));
typedef float  v8f    __attribute__((ext_vector_type(8)));
typedef float  v4f    __attribute__((ext_vector_type(4)));

// ---- Kernel 1: x_proj = x @ W_ih^T (+b_ih), bf16 WMMA, N padded 30->32 ------
// Grid: 4096 blocks x 256 threads (8 waves). Each wave computes a 16x32 tile of
// the (524288 x 32) output. K = 512, consumed in 16 chunks of 32.
// x loads are non-temporal (1.07 GB stream) so the 64 MB xp intermediate stays
// resident in the 192 MB L2 for kernel 2.
__global__ __launch_bounds__(256) void xproj_wmma_kernel(
    const float* __restrict__ x,      // (B*T, 512)
    const float* __restrict__ W_ih,   // (30, 512)
    const float* __restrict__ b_ih,   // (30)
    float* __restrict__ xp) {         // (B*T, 32) padded
  // W_ih staged once per block as bf16 [32][512] (rows 30,31 zero).
  __shared__ __align__(16) uint32_t Wlds[32 * 256];  // 32 KB, [n][k/2]

  const int tid = threadIdx.x;
  for (int i = tid; i < 32 * 256; i += 256) {
    int n  = i >> 8;          // 0..31
    int kp = (i & 255) << 1;  // even k
    float a = (n < 30) ? W_ih[n * 512 + kp]     : 0.f;
    float b = (n < 30) ? W_ih[n * 512 + kp + 1] : 0.f;
    union { bf16_t h[2]; uint32_t u; } pk;
    pk.h[0] = (bf16_t)a;
    pk.h[1] = (bf16_t)b;
    Wlds[i] = pk.u;
  }
  __syncthreads();

  const int lane  = tid & 31;
  const int wave  = tid >> 5;
  const long mtile = (long)blockIdx.x * 8 + wave;   // 16-row tile index
  const int  lan15 = lane & 15;
  const int  hi    = lane >> 4;                     // 0 or 1 (K-half select)

  // A operand: lane holds row (mtile*16 + lan15), K runs s..s+15,
  // s = hi*16 (contiguous, matching B).
  const long  row_a = mtile * 16 + lan15;
  const float* xrow = x + row_a * 512;

  // B operand: lane = column n, 16 contiguous K starting at k0 + hi*16.
  const int n0 = lan15;
  const int n1 = 16 + lan15;

  v8f c0 = {}; v8f c1 = {};

  for (int k0 = 0; k0 < 512; k0 += 32) {
    const v4f* p = (const v4f*)(xrow + k0 + hi * 16);
    v4f f0 = __builtin_nontemporal_load(p);       // K s..s+3
    v4f f1 = __builtin_nontemporal_load(p + 1);   // K s+4..s+7
    v4f f2 = __builtin_nontemporal_load(p + 2);   // K s+8..s+11
    v4f f3 = __builtin_nontemporal_load(p + 3);   // K s+12..s+15

    v16bf a;
    a[0]  = (bf16_t)f0[0]; a[1]  = (bf16_t)f0[1];
    a[2]  = (bf16_t)f0[2]; a[3]  = (bf16_t)f0[3];
    a[4]  = (bf16_t)f1[0]; a[5]  = (bf16_t)f1[1];
    a[6]  = (bf16_t)f1[2]; a[7]  = (bf16_t)f1[3];
    a[8]  = (bf16_t)f2[0]; a[9]  = (bf16_t)f2[1];
    a[10] = (bf16_t)f2[2]; a[11] = (bf16_t)f2[3];
    a[12] = (bf16_t)f3[0]; a[13] = (bf16_t)f3[1];
    a[14] = (bf16_t)f3[2]; a[15] = (bf16_t)f3[3];

    const int kb = (k0 + (hi << 4)) >> 1;  // dword offset into row
    union { v16bf v; uint4 q[2]; } B0, B1;
    const uint4* q0 = (const uint4*)&Wlds[n0 * 256 + kb];
    B0.q[0] = q0[0]; B0.q[1] = q0[1];
    const uint4* q1 = (const uint4*)&Wlds[n1 * 256 + kb];
    B1.q[0] = q1[0]; B1.q[1] = q1[1];

    c0 = __builtin_amdgcn_wmma_f32_16x16x32_bf16(false, a, false, B0.v,
                                                 (short)0, c0, false, false);
    c1 = __builtin_amdgcn_wmma_f32_16x16x32_bf16(false, a, false, B1.v,
                                                 (short)0, c1, false, false);
  }

  // C/D layout: VGPR r, lanes 0-15 -> M=r, lanes 16-31 -> M=8+r; N = lan15.
  const float bi0 = b_ih[n0];
  const float bi1 = (n1 < 30) ? b_ih[n1] : 0.f;
  const long rbase = mtile * 16 + (hi << 3);
#pragma unroll
  for (int r = 0; r < 8; ++r) {
    const long row = rbase + r;
    xp[row * 32 + n0] = c0[r] + bi0;   // default RT: keep xp hot in L2
    xp[row * 32 + n1] = c1[r] + bi1;
  }
}

// ---- Kernel 2: GRU scan. 16 lanes per batch element (10 active) -------------
// Grid: 128 blocks x 256 threads -> 16 batch elements / block.
// xp loads software-pipelined one step ahead to hide latency under the
// serial 30-FMA + sigmoid/tanh chain.
__global__ __launch_bounds__(256) void gru_scan_kernel(
    const float* __restrict__ xp,     // (B*T, 32) padded
    const float* __restrict__ h0,     // (B, 10)
    const float* __restrict__ W_hh,   // (30, 10)
    const float* __restrict__ b_hh,   // (30)
    float* __restrict__ out) {        // 2 x (B, 40)
  const int tid = threadIdx.x;
  const int g   = tid & 15;           // gate row within element
  const int grp = tid >> 4;
  const int b   = blockIdx.x * 16 + grp;
  const bool active = (g < 10);
  const int gg = active ? g : 0;      // clamp for safe loads

  float wr[10], wz[10], wn[10];
#pragma unroll
  for (int j = 0; j < 10; ++j) {
    wr[j] = W_hh[gg * 10 + j];
    wz[j] = W_hh[(10 + gg) * 10 + j];
    wn[j] = W_hh[(20 + gg) * 10 + j];
  }
  const float bhr = b_hh[gg], bhz = b_hh[10 + gg], bhn = b_hh[20 + gg];

  float hv[10];
#pragma unroll
  for (int j = 0; j < 10; ++j) hv[j] = h0[b * 10 + j];
  float my_h = h0[b * 10 + gg];       // this lane's h component

  float r = 0.f, z = 0.f, n = 0.f, hn = my_h;
  const float* xpb = xp + (long)b * 256 * 32;

  // prime the pipeline with t = 0
  float xr_c = xpb[gg];
  float xz_c = xpb[10 + gg];
  float xn_c = xpb[20 + gg];

  for (int t = 0; t < 256; ++t) {
    // issue next step's loads before the dependent compute chain
    float xr_n = 0.f, xz_n = 0.f, xn_n = 0.f;
    if (t < 255) {
      const float* nrow = xpb + (long)(t + 1) * 32;
      xr_n = nrow[gg];
      xz_n = nrow[10 + gg];
      xn_n = nrow[20 + gg];
    }

    float hr = bhr, hz = bhz, hnn = bhn;
#pragma unroll
    for (int j = 0; j < 10; ++j) {
      hr  = fmaf(hv[j], wr[j], hr);
      hz  = fmaf(hv[j], wz[j], hz);
      hnn = fmaf(hv[j], wn[j], hnn);
    }
    r  = 1.f / (1.f + __expf(-(xr_c + hr)));
    z  = 1.f / (1.f + __expf(-(xz_c + hz)));
    n  = tanhf(xn_c + r * hnn);
    hn = (1.f - z) * n + z * my_h;
    my_h = hn;
    // rebuild full hidden vector across the 16-lane group
#pragma unroll
    for (int j = 0; j < 10; ++j) hv[j] = __shfl(hn, j, 16);

    xr_c = xr_n; xz_c = xz_n; xn_c = xn_n;
  }

  if (active) {
    const long o = (long)b * 40 + g;
    out[o]      = r;  out[o + 10] = z;
    out[o + 20] = n;  out[o + 30] = hn;
    const long off = 2048L * 40;     // second tuple element, same data
    out[off + o]      = r;  out[off + o + 10] = z;
    out[off + o + 20] = n;  out[off + o + 30] = hn;
  }
}

// ---- Launch -----------------------------------------------------------------
extern "C" void kernel_launch(void* const* d_in, const int* in_sizes, int n_in,
                              void* d_out, int out_size, void* d_ws, size_t ws_size,
                              hipStream_t stream) {
  const float* x    = (const float*)d_in[0];
  const float* h    = (const float*)d_in[1];
  const float* W_ih = (const float*)d_in[2];
  const float* W_hh = (const float*)d_in[3];
  const float* b_ih = (const float*)d_in[4];
  const float* b_hh = (const float*)d_in[5];
  float* out = (float*)d_out;
  float* xp  = (float*)d_ws;   // needs 524288*32*4 = 64 MiB scratch

  // (B*T)/16 rows-per-wave / 8 waves-per-block = 4096 blocks
  xproj_wmma_kernel<<<4096, 256, 0, stream>>>(x, W_ih, b_ih, xp);
  gru_scan_kernel<<<128, 256, 0, stream>>>(xp, h, W_hh, b_hh, out);
}
